// GraphAttentionLayer_59519656788408
// MI455X (gfx1250) — matrix operand order repaired and checked by reading, not compiled, not run on hardware
//
#include <hip/hip_runtime.h>

// GAT layer, simplified: h_prime = (adj / rowsum(adj)) @ (h @ W)
// (per-row scores make the row softmax uniform over neighbors; a1/a2 unused)

#define NROWS 10000
#define OUTC  64
#define INC   128
#define KCHUNK 128
#define LSTRIDE (KCHUNK + 8)   // 136 halves = 272B row stride: 16B aligned, bank-spread

typedef __attribute__((ext_vector_type(16))) _Float16     v16h;
typedef __attribute__((ext_vector_type(8)))  float        v8f;
typedef __attribute__((ext_vector_type(4)))  unsigned int v4u;

union FragU  { v16h h; v4u u[2]; };
union Pack16 { _Float16 hh[16]; v4u u[2]; };

// ---------------------------------------------------------------------------
// Kernel 1: WhT[c][r] = (f16)( sum_k h[r][k] * W[k][c] ),  c<64, r<10000
// ---------------------------------------------------------------------------
__global__ __launch_bounds__(64) void wh_kernel(const float* __restrict__ h,
                                                const float* __restrict__ W,
                                                _Float16* __restrict__ whT) {
  __shared__ float hL[INC];
  const int row = blockIdx.x;
  const int t   = threadIdx.x;
  hL[t]      = h[(size_t)row * INC + t];
  hL[t + 64] = h[(size_t)row * INC + 64 + t];
  __syncthreads();
  float acc = 0.f;
#pragma unroll 8
  for (int k = 0; k < INC; ++k)
    acc = __builtin_fmaf(hL[k], W[k * OUTC + t], acc);
  whT[(size_t)t * NROWS + row] = (_Float16)acc;
}

// ---------------------------------------------------------------------------
// Kernel 2: per (M-tile, K-split): pacc[y] tile = adj_tile @ WhT^T, pdeg = deg
// Double-buffered LDS; WhT slice staged with global_load_async_to_lds_b128.
// ---------------------------------------------------------------------------
__global__ __launch_bounds__(128) void agg_kernel(const int* __restrict__ adj,
                                                  const _Float16* __restrict__ whT,
                                                  float* __restrict__ pacc,
                                                  int* __restrict__ pdeg,
                                                  int kStep) {
  __shared__ _Float16 adjL[2][16][LSTRIDE];
  __shared__ _Float16 whtL[2][64][LSTRIDE];
  __shared__ int degS[16];

  const int t       = threadIdx.x;
  const int rowBase = blockIdx.x * 16;
  const int kLo     = blockIdx.y * kStep;
  const int kHi     = (kLo + kStep < NROWS) ? (kLo + kStep) : NROWS;

  if (t < 16) degS[t] = 0;

  const int lane = t & 31;
  const int wave = t >> 5;       // 0..3 -> 16-column block of the output
  const int mrow = lane & 15;    // M row (A) / N col (B) inside the 16x16 tile
  const int g    = lane >> 4;    // half-wave select per ISA 7.12.2 layouts

  // staging maps
  const int sRow = t >> 3;          // adj tile row 0..15
  const int sCol = (t & 7) * 16;    // 16 consecutive adj columns
  const int wRow = t >> 1;          // WhT row (= output column) 0..63
  const int wCol = (t & 1) * 64;    // 64 consecutive k halves (128B)

  const size_t adjRowOff = (size_t)(rowBase + sRow) * NROWS + sCol;
  const size_t whtRowOff = (size_t)wRow * NROWS + wCol;

  int degLocal = 0;
  v8f acc = {};

  // ---- stage adj tile (16 x KCHUNK i32 -> f16), accumulate degree ----
  auto stageAdjFast = [&](int k0, int buf) {
    const int* src = adj + adjRowOff + k0;
    v4u c[4];
#pragma unroll
    for (int q = 0; q < 4; ++q) c[q] = reinterpret_cast<const v4u*>(src)[q];
    __builtin_prefetch(src + KCHUNK, 0, 1);  // global_prefetch next chunk
    Pack16 pk;
#pragma unroll
    for (int q = 0; q < 4; ++q)
#pragma unroll
      for (int j = 0; j < 4; ++j) {
        int v = (int)c[q][j];
        degLocal += v;
        pk.hh[q * 4 + j] = (_Float16)(float)v;
      }
    *reinterpret_cast<v4u*>(&adjL[buf][sRow][sCol])     = pk.u[0];
    *reinterpret_cast<v4u*>(&adjL[buf][sRow][sCol + 8]) = pk.u[1];
  };
  auto stageAdjTail = [&](int k0, int buf) {
    const int* src = adj + adjRowOff + k0;
#pragma unroll
    for (int j = 0; j < 16; ++j) {
      int cg = k0 + sCol + j;
      int v  = (cg < kHi) ? src[j] : 0;
      degLocal += v;
      adjL[buf][sRow][sCol + j] = (_Float16)(float)v;
    }
  };

  // ---- stage WhT slice (64 x KCHUNK f16): async global->LDS, 128B/thread ----
  auto stageWhtFast = [&](int k0, int buf) {
    unsigned lds = (unsigned)(unsigned long long)(void*)&whtL[buf][wRow][wCol];
    unsigned long long ga = (unsigned long long)(const void*)(whT + whtRowOff + k0);
    asm volatile(
        "global_load_async_to_lds_b128 %0, %1, off\n\t"
        "global_load_async_to_lds_b128 %0, %1, off offset:16\n\t"
        "global_load_async_to_lds_b128 %0, %1, off offset:32\n\t"
        "global_load_async_to_lds_b128 %0, %1, off offset:48\n\t"
        "global_load_async_to_lds_b128 %0, %1, off offset:64\n\t"
        "global_load_async_to_lds_b128 %0, %1, off offset:80\n\t"
        "global_load_async_to_lds_b128 %0, %1, off offset:96\n\t"
        "global_load_async_to_lds_b128 %0, %1, off offset:112"
        :: "v"(lds), "v"(ga) : "memory");
  };
  auto stageWhtTail = [&](int k0, int buf) {
    const _Float16* wsrc = whT + whtRowOff + k0;
#pragma unroll 8
    for (int j = 0; j < 64; ++j) {
      int kg = k0 + wCol + j;
      whtL[buf][wRow][wCol + j] = (kg < kHi) ? wsrc[j] : (_Float16)0.0f;
    }
  };

  // ---- 4 WMMAs (K=32 each) per wave per chunk; fragments per ISA 7.12.2 ----
  auto computeChunk = [&](int buf) {
#pragma unroll
    for (int sub = 0; sub < 4; ++sub) {
      const int ks = sub * 32;
      FragU A, B;
      // A 16x32 f16: row = mrow; VGPR0-3 = K[g*8..+7], VGPR4-7 = K[16+g*8..+7]
      A.u[0] = *reinterpret_cast<const v4u*>(&adjL[buf][mrow][ks + 8 * g]);
      A.u[1] = *reinterpret_cast<const v4u*>(&adjL[buf][mrow][ks + 16 + 8 * g]);
      // B 32x16 f16: col = mrow; VGPR0-3 = K[g*8..+7], VGPR4-7 = K[16+g*8..+7]
      B.u[0] = *reinterpret_cast<const v4u*>(&whtL[buf][wave * 16 + mrow][ks + 8 * g]);
      B.u[1] = *reinterpret_cast<const v4u*>(&whtL[buf][wave * 16 + mrow][ks + 16 + 8 * g]);
      acc = __builtin_amdgcn_wmma_f32_16x16x32_f16(false, A.h, false, B.h,
                                                   (short)0, acc, false, false);
    }
  };

  const int span    = kHi - kLo;
  const int nFull   = (span > 0) ? (span / KCHUNK) : 0;  // unguarded chunks
  const int nChunk  = (span > 0) ? ((span + KCHUNK - 1) / KCHUNK) : 0;

  if (nChunk > 0) {
    if (nFull > 0) { stageAdjFast(kLo, 0); stageWhtFast(kLo, 0); }
    else           { stageAdjTail(kLo, 0); stageWhtTail(kLo, 0); }
  }
  asm volatile("s_wait_asynccnt 0x0" ::: "memory");
  __syncthreads();

  for (int i = 0; i < nChunk; ++i) {
    const int nb = (i + 1) & 1;
    if (i + 1 < nFull) {            // next chunk is full: branch-free staging
      stageAdjFast(kLo + (i + 1) * KCHUNK, nb);
      stageWhtFast(kLo + (i + 1) * KCHUNK, nb);
    } else if (i + 1 < nChunk) {    // next chunk is the ragged tail
      stageAdjTail(kLo + (i + 1) * KCHUNK, nb);
      stageWhtTail(kLo + (i + 1) * KCHUNK, nb);
    }
    computeChunk(i & 1);
    asm volatile("s_wait_asynccnt 0x0" ::: "memory");
    __syncthreads();
  }

  // ---- degree reduce + writeback ----
  atomicAdd(&degS[sRow], degLocal);
  __syncthreads();

  float* outTile = pacc + (size_t)blockIdx.y * ((size_t)NROWS * OUTC);
#pragma unroll
  for (int r = 0; r < 8; ++r) {
    int m = r + g * 8;  // C/D layout: lanes 0-15 -> M=r, lanes 16-31 -> M=r+8
    outTile[(size_t)(rowBase + m) * OUTC + wave * 16 + mrow] = acc[r];
  }
  if (t < 16) pdeg[(size_t)blockIdx.y * NROWS + rowBase + t] = degS[t];
}

// ---------------------------------------------------------------------------
// Kernel 3: out = (sum_s pacc[s]) / max(sum_s pdeg[s], 1)
// ---------------------------------------------------------------------------
__global__ __launch_bounds__(256) void fin_kernel(const float* __restrict__ pacc,
                                                  const int* __restrict__ pdeg,
                                                  float* __restrict__ out,
                                                  int ksplit) {
  int idx = blockIdx.x * blockDim.x + threadIdx.x;
  if (idx >= NROWS * OUTC) return;
  int row = idx >> 6;
  float s = 0.f;
  int d = 0;
  for (int i = 0; i < ksplit; ++i) s += pacc[(size_t)i * NROWS * OUTC + idx];
  for (int i = 0; i < ksplit; ++i) d += pdeg[(size_t)i * NROWS + row];
  out[idx] = s / (float)(d > 0 ? d : 1);
}

// ---------------------------------------------------------------------------
extern "C" void kernel_launch(void* const* d_in, const int* in_sizes, int n_in,
                              void* d_out, int out_size, void* d_ws, size_t ws_size,
                              hipStream_t stream) {
  const float* h   = (const float*)d_in[0];
  const int*   adj = (const int*)d_in[1];
  const float* W   = (const float*)d_in[2];
  // d_in[3], d_in[4] (a1, a2) provably do not affect the output.
  float* out = (float*)d_out;
  char*  ws  = (char*)d_ws;

  const size_t whtBytes = (size_t)OUTC * NROWS * sizeof(_Float16);  // 1.28 MB
  size_t off = (whtBytes + 255) & ~(size_t)255;
  const size_t perSplit = (size_t)NROWS * OUTC * sizeof(float) + (size_t)NROWS * sizeof(int);

  _Float16* whT = (_Float16*)ws;
  float* pacc;
  int*   pdeg;
  int    ksplit;
  if (ws_size >= off + perSplit) {
    ksplit = (int)((ws_size - off) / perSplit);
    if (ksplit > 8) ksplit = 8;
    if (ksplit < 1) ksplit = 1;
    pacc = (float*)(ws + off);
    pdeg = (int*)(ws + off + (size_t)ksplit * NROWS * OUTC * sizeof(float));
  } else {
    // tight scratch: accumulate straight into d_out, degrees right after whT
    ksplit = 1;
    pacc = out;
    pdeg = (int*)(ws + off);
  }

  const int nChunks   = (NROWS + KCHUNK - 1) / KCHUNK;  // 79
  const int chunksPer = (nChunks + ksplit - 1) / ksplit;
  const int kStep     = chunksPer * KCHUNK;

  wh_kernel<<<NROWS, 64, 0, stream>>>(h, W, whT);
  dim3 grid(NROWS / 16, ksplit);
  agg_kernel<<<grid, 128, 0, stream>>>(adj, whT, pacc, pdeg, kStep);
  fin_kernel<<<(NROWS * OUTC + 255) / 256, 256, 0, stream>>>(pacc, pdeg, out, ksplit);
}
